// rgcn_bert_69398081569028
// MI455X (gfx1250) — matrix operand order repaired and checked by reading, not compiled, not run on hardware
//
#include <hip/hip_runtime.h>
#include <math.h>

// Problem constants (from reference)
#define N_NODES 100000
#define EMB     256
#define HID     64
#define N_REL   5
#define N_EDGES 3200000
#define BATCH   16384
#define NREL6   6          // 5 relations + root as pseudo-relation 5
#define NRSEG   (N_NODES * N_REL)

typedef __attribute__((ext_vector_type(16))) __bf16 v16bf;
typedef __attribute__((ext_vector_type(8)))  __bf16 v8bf;
typedef __attribute__((ext_vector_type(8)))  float  v8f;

// ---------------------------------------------------------------- utilities

__global__ void zero_kernel(unsigned int* __restrict__ p, long long n) {
  long long i = (long long)blockIdx.x * blockDim.x + threadIdx.x;
  if (i < n) p[i] = 0u;
}

__global__ void cast_bf16_kernel(const float* __restrict__ in,
                                 __bf16* __restrict__ out, long long n) {
  long long i = (long long)blockIdx.x * blockDim.x + threadIdx.x;
  if (i < n) out[i] = (__bf16)in[i];
}

// WT[r][o][k]  (r: 0..4 = W[r], r==5 = root), contiguous in k for WMMA B loads
__global__ void prep_wt_kernel(const float* __restrict__ W,
                               const float* __restrict__ root,
                               __bf16* __restrict__ WT, int K) {
  int gid = blockIdx.x * blockDim.x + threadIdx.x;
  if (gid >= NREL6 * 64 * K) return;
  int k = gid % K;
  int o = (gid / K) & 63;
  int r = gid / (K * 64);
  float v = (r < N_REL) ? W[((size_t)r * K + k) * 64 + o]
                        : root[(size_t)k * 64 + o];
  WT[gid] = (__bf16)v;
}

// ------------------------------------------------------ degree counts / inv

__global__ void count_kernel(const int* __restrict__ dst,
                             const int* __restrict__ et,
                             unsigned int* __restrict__ cnt) {
  int e = blockIdx.x * blockDim.x + threadIdx.x;
  if (e < N_EDGES)
    atomicAdd(&cnt[(size_t)dst[e] * N_REL + et[e]], 1u);
}

__global__ void inv_kernel(const unsigned int* __restrict__ cnt,
                           float* __restrict__ inv, int n) {
  int i = blockIdx.x * blockDim.x + threadIdx.x;
  if (i < n) inv[i] = 1.0f / fmaxf((float)cnt[i], 1.0f);
}

// ---------------------------------------------------- bf16 WMMA batched GEMM
// H[r][n][o] = sum_k X[n][k] * WT[r][o][k]   for all 6 relations at once.
// 8 waves/block = 2 row-tiles x 4 col-tiles; each wave accumulates 6
// relations so each A fragment is loaded once and feeds 6 WMMAs.
// K is compile-time; full unroll + load-all-then-mma-all per k-step lets the
// scheduler clause 12 B loads into distinct register pairs and overlap them
// with the WMMAs (partial s_wait_loadcnt instead of wait-0 per WMMA).

template <int K>
__global__ __launch_bounds__(256)
void gemm_all_rel_kernel(const __bf16* __restrict__ X,
                         const __bf16* __restrict__ WT,  // [6][64][K]
                         float* __restrict__ H, int Nrows) {
  const int lane = threadIdx.x & 31;
  const int wave = threadIdx.x >> 5;           // 0..7
  const int t    = wave & 3;                   // col tile (n0 = t*16)
  const int m0   = (blockIdx.x * 2 + (wave >> 2)) * 16;
  if (m0 >= Nrows) return;                     // wave-uniform: EXEC all-1s

  const int half = lane >> 4;                  // 0: lanes 0-15, 1: 16-31
  const int lid  = lane & 15;
  const size_t relStride = (size_t)64 * K;

  // A fragment per-lane base: elements {k0+16h..+15} (contiguous)
  const __bf16* Xrow  = X + (size_t)(m0 + lid) * K + half * 16;
  // B fragment per-lane base: col = t*16+lid, K = k0+16h..+15 (contiguous 32B)
  const __bf16* Wbase = WT + (size_t)(t * 16 + lid) * K + half * 16;

  v8f acc[NREL6];
#pragma unroll
  for (int r = 0; r < NREL6; ++r) acc[r] = v8f{};

#pragma unroll
  for (int k0 = 0; k0 < K; k0 += 32) {
    union { v16bf v; v8bf h[2]; } au;
    au.h[0] = *(const v8bf*)(Xrow + k0);        // lands in low 4 VGPRs
    au.h[1] = *(const v8bf*)(Xrow + k0 + 8);    // lands in high 4 VGPRs

    // issue all 6 B-fragment loads first (distinct register pairs)
    v16bf b[NREL6];
#pragma unroll
    for (int r = 0; r < NREL6; ++r)
      b[r] = *(const v16bf*)(Wbase + r * relStride + k0);

    // then the 6 WMMAs; loads for later r remain in flight
#pragma unroll
    for (int r = 0; r < NREL6; ++r)
      acc[r] = __builtin_amdgcn_wmma_f32_16x16x32_bf16(
          false, au.v, false, b[r], (short)0, acc[r], false, false);
  }

  // C/D layout: VGPR i -> M = m0 + i + 8*half, N = t*16 + lid
#pragma unroll
  for (int r = 0; r < NREL6; ++r) {
    float* p = H + ((size_t)r * Nrows + m0 + half * 8) * 64 + t * 16 + lid;
#pragma unroll
    for (int i = 0; i < 8; ++i) p[(size_t)i * 64] = acc[r][i];
  }
}

// ----------------------------------------------- edge scatter (mean weights)
// 16 threads per edge, 4 floats each: AGG[dst] += inv[dst,rel] * H[rel][src]

__global__ void scatter_kernel(const float* __restrict__ H,
                               const int* __restrict__ src,
                               const int* __restrict__ dst,
                               const int* __restrict__ et,
                               const float* __restrict__ inv,
                               float* __restrict__ AGG) {
  long long gid = (long long)blockIdx.x * blockDim.x + threadIdx.x;
  long long e = gid >> 4;
  if (e >= N_EDGES) return;
  int sub = (int)(gid & 15);
  int r = et[e], s = src[e], d = dst[e];
  float w = inv[(size_t)d * N_REL + r];
  const float4 v = *(const float4*)(H + ((size_t)r * N_NODES + s) * 64 + sub * 4);
  float* out = AGG + (size_t)d * 64 + sub * 4;
  atomicAdd(out + 0, w * v.x);
  atomicAdd(out + 1, w * v.y);
  atomicAdd(out + 2, w * v.z);
  atomicAdd(out + 3, w * v.w);
}

// ------------------------------------------------------------- combine ops

__global__ void combine_relu_bf16_kernel(const float* __restrict__ AGG,
                                         const float* __restrict__ Hroot,
                                         const float* __restrict__ b,
                                         __bf16* __restrict__ xout,
                                         long long n) {
  long long i = (long long)blockIdx.x * blockDim.x + threadIdx.x;
  if (i >= n) return;
  int o = (int)(i & 63);
  float v = AGG[i] + Hroot[i] + b[o];
  xout[i] = (__bf16)fmaxf(v, 0.0f);
}

__global__ void combine_f32_kernel(const float* __restrict__ AGG,
                                   const float* __restrict__ Hroot,
                                   const float* __restrict__ b,
                                   float* __restrict__ nodes,
                                   long long n) {
  long long i = (long long)blockIdx.x * blockDim.x + threadIdx.x;
  if (i >= n) return;
  int o = (int)(i & 63);
  nodes[i] = AGG[i] + Hroot[i] + b[o];
}

// ------------------------------------------------------- merge MLP + loss

__global__ void merge_fc1_kernel(const float* __restrict__ nodes,
                                 const int* __restrict__ bill,
                                 const int* __restrict__ u1,
                                 const int* __restrict__ u2,
                                 const float* __restrict__ w1,   // [128,64]
                                 const float* __restrict__ b1,   // [64]
                                 float* __restrict__ hpos,
                                 float* __restrict__ hneg) {
  int gid = blockIdx.x * blockDim.x + threadIdx.x;   // 2 * 16384 * 64
  int o    = gid & 63;
  int s    = (gid >> 6) & (BATCH - 1);
  int pair = gid >> 20;
  const float* xb = nodes + (size_t)bill[s] * 64;
  const float* xu = nodes + (size_t)(pair ? u2[s] : u1[s]) * 64;
  float acc = b1[o];
#pragma unroll 4
  for (int i = 0; i < 64; ++i) acc += xb[i] * w1[(size_t)i * 64 + o];
#pragma unroll 4
  for (int i = 0; i < 64; ++i) acc += xu[i] * w1[(size_t)(64 + i) * 64 + o];
  (pair ? hneg : hpos)[(size_t)s * 64 + o] = fmaxf(acc, 0.0f);
}

__global__ void fc2_loss_kernel(const float* __restrict__ hpos,
                                const float* __restrict__ hneg,
                                const float* __restrict__ w2,   // [64]
                                const float* __restrict__ b2,   // [1]
                                float* __restrict__ red) {
  int gid = blockIdx.x * blockDim.x + threadIdx.x;   // 2 * 16384
  int pair = gid >> 14;
  int s    = gid & (BATCH - 1);
  const float* h = (pair ? hneg : hpos) + (size_t)s * 64;
  float z = b2[0];
#pragma unroll 4
  for (int i = 0; i < 64; ++i) z += h[i] * w2[i];
  float t = pair ? 0.0f : 1.0f;
  float bce = fmaxf(z, 0.0f) - z * t + log1pf(expf(-fabsf(z)));
  float pred = (z > 0.0f) ? 1.0f : 0.0f;
  float corr = (pred == t) ? 1.0f : 0.0f;
  // wave reduce then 2 atomics per wave
  for (int off = 16; off > 0; off >>= 1) {
    bce  += __shfl_down(bce, off, 32);
    corr += __shfl_down(corr, off, 32);
  }
  if ((threadIdx.x & 31) == 0) {
    atomicAdd(&red[0], bce);
    atomicAdd(&red[1], corr);
  }
}

__global__ void finalize_kernel(const float* __restrict__ red,
                                float* __restrict__ out) {
  out[0] = red[0] / (float)(2 * BATCH);
  out[1] = red[1] / (float)(2 * BATCH);
}

// ------------------------------------------------------------------ launch

extern "C" void kernel_launch(void* const* d_in, const int* in_sizes, int n_in,
                              void* d_out, int out_size, void* d_ws, size_t ws_size,
                              hipStream_t stream) {
  const int*   bill  = (const int*)d_in[0];
  const int*   u1    = (const int*)d_in[1];
  const int*   u2    = (const int*)d_in[2];
  const int*   ei    = (const int*)d_in[3];
  const int*   src   = ei;
  const int*   dst   = ei + N_EDGES;
  const int*   et    = (const int*)d_in[4];
  const float* x     = (const float*)d_in[5];
  const float* W1    = (const float*)d_in[6];
  const float* root1 = (const float*)d_in[7];
  const float* b1    = (const float*)d_in[8];
  const float* W2    = (const float*)d_in[9];
  const float* root2 = (const float*)d_in[10];
  const float* b2    = (const float*)d_in[11];
  const float* m1w   = (const float*)d_in[12];
  const float* m1b   = (const float*)d_in[13];
  const float* m2w   = (const float*)d_in[14];
  const float* m2b   = (const float*)d_in[15];

  // workspace carve-up (256B aligned)
  char* base = (char*)d_ws;
  size_t off = 0;
  auto take = [&](size_t bytes) -> char* {
    char* p = base + off;
    off = (off + bytes + 255) & ~(size_t)255;
    return p;
  };
  __bf16* xb    = (__bf16*)take((size_t)N_NODES * EMB * 2);       // 51.2 MB
  __bf16* wt1   = (__bf16*)take((size_t)NREL6 * 64 * EMB * 2);
  __bf16* wt2   = (__bf16*)take((size_t)NREL6 * 64 * HID * 2);
  __bf16* x1b   = (__bf16*)take((size_t)N_NODES * HID * 2);       // 12.8 MB
  float*  H     = (float*)take((size_t)NREL6 * N_NODES * 64 * 4); // 153.6 MB
  unsigned int* cnt = (unsigned int*)take((size_t)NRSEG * 4);
  float*  inv   = (float*)take((size_t)NRSEG * 4);
  float*  AGG   = (float*)take((size_t)N_NODES * 64 * 4);         // 25.6 MB
  float*  nodes = (float*)take((size_t)N_NODES * 64 * 4);         // 25.6 MB
  float*  hpos  = (float*)take((size_t)BATCH * 64 * 4);
  float*  hneg  = (float*)take((size_t)BATCH * 64 * 4);
  float*  red   = (float*)take(256);
  (void)ws_size; (void)in_sizes; (void)n_in; (void)out_size;

  const long long NH    = (long long)N_NODES * 64;       // 6.4M
  const float*    Hroot = H + (size_t)N_REL * N_NODES * 64;
  const int gemmBlocks  = ((N_NODES / 16) + 1) / 2;      // 3125

  // degree counts (graph fixed for both layers)
  zero_kernel<<<(NRSEG + 255) / 256, 256, 0, stream>>>(cnt, NRSEG);
  count_kernel<<<N_EDGES / 256, 256, 0, stream>>>(dst, et, cnt);
  inv_kernel<<<(NRSEG + 255) / 256, 256, 0, stream>>>(cnt, inv, NRSEG);

  // bf16 inputs/weights
  cast_bf16_kernel<<<(int)(((long long)N_NODES * EMB) / 256), 256, 0, stream>>>(
      x, xb, (long long)N_NODES * EMB);
  prep_wt_kernel<<<(NREL6 * 64 * EMB + 255) / 256, 256, 0, stream>>>(W1, root1, wt1, EMB);
  prep_wt_kernel<<<(NREL6 * 64 * HID + 255) / 256, 256, 0, stream>>>(W2, root2, wt2, HID);

  // ---- layer 1: transform (WMMA) -> scatter-mean -> +root +bias, relu, bf16
  gemm_all_rel_kernel<EMB><<<gemmBlocks, 256, 0, stream>>>(xb, wt1, H, N_NODES);
  zero_kernel<<<(int)(NH / 256), 256, 0, stream>>>((unsigned int*)AGG, NH);
  scatter_kernel<<<(int)(((long long)N_EDGES * 16) / 256), 256, 0, stream>>>(
      H, src, dst, et, inv, AGG);
  combine_relu_bf16_kernel<<<(int)(NH / 256), 256, 0, stream>>>(AGG, Hroot, b1, x1b, NH);

  // ---- layer 2
  gemm_all_rel_kernel<HID><<<gemmBlocks, 256, 0, stream>>>(x1b, wt2, H, N_NODES);
  zero_kernel<<<(int)(NH / 256), 256, 0, stream>>>((unsigned int*)AGG, NH);
  scatter_kernel<<<(int)(((long long)N_EDGES * 16) / 256), 256, 0, stream>>>(
      H, src, dst, et, inv, AGG);
  combine_f32_kernel<<<(int)(NH / 256), 256, 0, stream>>>(AGG, Hroot, b2, nodes, NH);

  // ---- merge MLP + BCE/accuracy
  merge_fc1_kernel<<<(2 * BATCH * 64) / 256, 256, 0, stream>>>(
      nodes, bill, u1, u2, m1w, m1b, hpos, hneg);
  zero_kernel<<<1, 256, 0, stream>>>((unsigned int*)red, 2);
  fc2_loss_kernel<<<(2 * BATCH) / 256, 256, 0, stream>>>(hpos, hneg, m2w, m2b, red);
  finalize_kernel<<<1, 1, 0, stream>>>(red, (float*)d_out);
}
